// SegMambaBackbone2D_58583353917906
// MI455X (gfx1250) — compile-verified, matrix operand order checked by reading, not run on hardware
//
#include <hip/hip_runtime.h>
#include <hip/hip_bf16.h>

// ---------------------------------------------------------------------------
// CDNA5 (gfx1250) types
// ---------------------------------------------------------------------------
typedef __bf16 bf16_t;
typedef bf16_t v16bf __attribute__((ext_vector_type(16)));
typedef float  v8f   __attribute__((ext_vector_type(8)));
typedef unsigned int u32x4 __attribute__((ext_vector_type(4)));
typedef unsigned int u32x8 __attribute__((ext_vector_type(8)));

#define ACT_NONE 0
#define ACT_RELU 1
#define ACT_GELU 2
#define ACT_SILU 3
#define ACT_SOFTPLUS 4

__device__ __forceinline__ float act_apply(float v, int act) {
  switch (act) {
    case ACT_RELU:     return v > 0.f ? v : 0.f;
    case ACT_GELU:     return 0.5f * v * (1.f + erff(v * 0.70710678118654752f));
    case ACT_SILU:     return v / (1.f + expf(-v));
    case ACT_SOFTPLUS: return (v > 20.f) ? v : log1pf(expf(v));
    default:           return v;
  }
}

// ---------------------------------------------------------------------------
// TDM: DMA a 64-row x 32-col f32 tile (rows row0.., cols col0.. of a
// [nrows,ncols] row-major matrix with row stride ldsrc) into LDS at `lds`.
// Ragged edges are zero-filled by the D# tensor_dim clamping (ISA 8.2).
// ---------------------------------------------------------------------------
__device__ __forceinline__ void tdm_load_tile64x32(const float* src, int row0, int col0,
                                                   int nrows, int ncols, int ldsrc,
                                                   void* lds) {
#if defined(__HIP_DEVICE_COMPILE__)
  unsigned ldsoff = (unsigned)(unsigned long long)lds;
  unsigned long long ga =
      (unsigned long long)(const void*)(src + (long long)row0 * ldsrc + col0);
  int td0 = ncols - col0; if (td0 < 0) td0 = 0;   // valid cols from tile start
  int td1 = nrows - row0; if (td1 < 0) td1 = 0;   // valid rows from tile start
  u32x4 g0;
  g0[0] = 1u;                                     // count=1, user mode
  g0[1] = ldsoff;                                 // lds_addr
  g0[2] = (unsigned)ga;                           // global_addr[31:0]
  g0[3] = ((unsigned)(ga >> 32) & 0x01FFFFFFu) | (2u << 30);  // addr[56:32], type=2
  u32x8 g1;
  g1[0] = 2u << 16;                               // wg_mask=0, data_size=4B
  g1[1] = ((unsigned)td0 & 0xFFFFu) << 16;        // tensor_dim0[15:0] @bit48
  g1[2] = (((unsigned)td0 >> 16) & 0xFFFFu) |
          (((unsigned)td1 & 0xFFFFu) << 16);      // dim0 hi | tensor_dim1 lo
  g1[3] = (((unsigned)td1 >> 16) & 0xFFFFu) |
          (32u << 16);                            // dim1 hi | tile_dim0=32
  g1[4] = 64u;                                    // tile_dim1=64, tile_dim2=0
  g1[5] = (unsigned)ldsrc;                        // tensor_dim0_stride lo
  g1[6] = 0u;
  g1[7] = 0u;
  asm volatile("tensor_load_to_lds %0, %1" :: "s"(g0), "s"(g1) : "memory");
#else
  (void)src; (void)row0; (void)col0; (void)nrows; (void)ncols; (void)ldsrc; (void)lds;
#endif
}

template <int N>
__device__ __forceinline__ void wait_tensorcnt() {
#if defined(__HIP_DEVICE_COMPILE__)
  __builtin_amdgcn_s_wait_tensorcnt(N);   // immediate must be a constant
#endif
}

// ---------------------------------------------------------------------------
// WMMA GEMM:  C[M,N] = act(A[M,K] * W[N,K]^T + bias) (+resid)
//   A row-major with row stride lda, W row-major (out_features x in_features).
//   256 threads = 8 waves; block tile 64(M) x 64(N); each wave owns 16x32
//   (two 16x16 C tiles sharing one A fragment) -> 2 v_wmma per k-step.
//   Both A and W tiles are streamed by the Tensor Data Mover into
//   double-buffered LDS: wave0 drives A tiles, wave1 drives W tiles, and
//   `s_wait_tensorcnt 1` after issuing tile i+1 guarantees tile i landed
//   (TDM ops from one wave complete in order) -> DMA overlaps WMMA.
// ---------------------------------------------------------------------------
__global__ __launch_bounds__(256) void k_gemm(
    const float* __restrict__ A, int lda,
    const float* __restrict__ W, int ldw,
    const float* __restrict__ bias,
    const float* __restrict__ resid,
    float* __restrict__ C,
    int M, int N, int K, int act) {
  __shared__ float sA[2][64 * 32];
  __shared__ float sB[2][64 * 32];

  const int tid  = threadIdx.x;
  const int wave = tid >> 5;
  const int lane = tid & 31;
  const int mw   = wave & 3;    // M sub-tile 0..3
  const int nw   = wave >> 2;   // N half 0..1
  const int m0   = blockIdx.y * 64;
  const int n0   = blockIdx.x * 64;
  const int hl   = lane >> 4;   // lane half
  const int lm   = lane & 15;

  v8f acc0 = {};
  v8f acc1 = {};

  const int nk = (K + 31) >> 5;

  // pre-issue tile 0
  if (wave == 0)      tdm_load_tile64x32(A, m0, 0, M, K, lda, &sA[0][0]);
  else if (wave == 1) tdm_load_tile64x32(W, n0, 0, N, K, ldw, &sB[0][0]);

  for (int i = 0; i < nk; ++i) {
    const int cur = i & 1;
    if (i + 1 < nk) {
      const int k1 = (i + 1) << 5;
      if (wave == 0)      tdm_load_tile64x32(A, m0, k1, M, K, lda, &sA[cur ^ 1][0]);
      else if (wave == 1) tdm_load_tile64x32(W, n0, k1, N, K, ldw, &sB[cur ^ 1][0]);
      if (wave <= 1) wait_tensorcnt<1>();   // tile i complete, tile i+1 in flight
    } else {
      if (wave <= 1) wait_tensorcnt<0>();
    }
    __syncthreads();   // LDS tile i visible to all waves

    // --- build bf16 fragments (ISA 7.12.2 16-bit layouts) ---
    // lanes 0-15: VGPR0-3 K=0..7,  VGPR4-7 K=16..23
    // lanes16-31: VGPR0-3 K=8..15, VGPR4-7 K=24..31
    v16bf afr, bfr0, bfr1;
    {
      const float* pa  = &sA[cur][(mw * 16 + lm) * 32 + hl * 8];
      const float* pb0 = &sB[cur][(nw * 32 + lm) * 32 + hl * 8];
      const float* pb1 = pb0 + 16 * 32;
#pragma unroll
      for (int j = 0; j < 8; ++j) {
        afr[j]      = (bf16_t)pa[j];
        afr[j + 8]  = (bf16_t)pa[16 + j];
        bfr0[j]     = (bf16_t)pb0[j];
        bfr0[j + 8] = (bf16_t)pb0[16 + j];
        bfr1[j]     = (bf16_t)pb1[j];
        bfr1[j + 8] = (bf16_t)pb1[16 + j];
      }
    }
#if defined(__HIP_DEVICE_COMPILE__)
    acc0 = __builtin_amdgcn_wmma_f32_16x16x32_bf16(false, afr, false, bfr0,
                                                   (short)0, acc0, false, false);
    acc1 = __builtin_amdgcn_wmma_f32_16x16x32_bf16(false, afr, false, bfr1,
                                                   (short)0, acc1, false, false);
#endif
    __syncthreads();   // done reading buf[cur]; it may be re-filled at i+2
  }

  // --- epilogue: C layout VGPR v -> row m0+mw*16+hl*8+v ---
  const int col0 = n0 + nw * 32 + lm;
  const int col1 = col0 + 16;
#pragma unroll
  for (int v = 0; v < 8; ++v) {
    int row = m0 + mw * 16 + hl * 8 + v;
    if (row < M) {
      if (col0 < N) {
        float val = acc0[v];
        if (bias)  val += bias[col0];
        val = act_apply(val, act);
        long long o = (long long)row * N + col0;
        if (resid) val += resid[o];
        C[o] = val;
      }
      if (col1 < N) {
        float val = acc1[v];
        if (bias)  val += bias[col1];
        val = act_apply(val, act);
        long long o = (long long)row * N + col1;
        if (resid) val += resid[o];
        C[o] = val;
      }
    }
  }
}

// ---------------------------------------------------------------------------
// im2col (NCHW -> [B*OH*OW, Cin*KH*KW])
// ---------------------------------------------------------------------------
__global__ void k_im2col(const float* __restrict__ x, float* __restrict__ col,
                         int Bt, int Cin, int H, int W, int KH, int KW,
                         int stride, int pad, int OH, int OW) {
  int K = Cin * KH * KW;
  long long total = (long long)Bt * OH * OW * K;
  long long idx = (long long)blockIdx.x * blockDim.x + threadIdx.x;
  if (idx >= total) return;
  int k = (int)(idx % K);
  long long m = idx / K;
  int ow = (int)(m % OW);
  int oh = (int)((m / OW) % OH);
  int b  = (int)(m / ((long long)OW * OH));
  int kw = k % KW, kh = (k / KW) % KH, ci = k / (KW * KH);
  int ih = oh * stride - pad + kh;
  int iw = ow * stride - pad + kw;
  float v = 0.f;
  if (ih >= 0 && ih < H && iw >= 0 && iw < W)
    v = x[(((long long)b * Cin + ci) * H + ih) * W + iw];
  col[idx] = v;
}

// ---------------------------------------------------------------------------
// layout / elementwise kernels
// ---------------------------------------------------------------------------
// (B,L,C) row-major -> (B,C,L), optional elementwise add of `addp` (B,C,L)
__global__ void k_blc2bcl(const float* __restrict__ in, const float* __restrict__ addp,
                          float* __restrict__ out, int Bt, int L, int C) {
  long long tot = (long long)Bt * C * L;
  long long i = (long long)blockIdx.x * blockDim.x + threadIdx.x;
  if (i >= tot) return;
  int l = (int)(i % L);
  long long t = i / L;
  int c = (int)(t % C);
  int b = (int)(t / C);
  float v = in[((long long)b * L + l) * C + c];
  if (addp) v += addp[i];
  out[i] = v;
}

// (B,C,L) -> (B,L,C)
__global__ void k_bcl2blc(const float* __restrict__ in, float* __restrict__ out,
                          int Bt, int L, int C) {
  long long tot = (long long)Bt * L * C;
  long long i = (long long)blockIdx.x * blockDim.x + threadIdx.x;
  if (i >= tot) return;
  int c = (int)(i % C);
  long long t = i / C;
  int l = (int)(t % L);
  int b = (int)(t / L);
  out[i] = in[((long long)b * C + c) * L + l];
}

__global__ void k_add2(const float* a, const float* b, float* o, long long n) {
  long long i = (long long)blockIdx.x * blockDim.x + threadIdx.x;
  if (i < n) o[i] = a[i] + b[i];
}
__global__ void k_add3(const float* a, const float* b, const float* c, float* o, long long n) {
  long long i = (long long)blockIdx.x * blockDim.x + threadIdx.x;
  if (i < n) o[i] = a[i] + b[i] + c[i];
}

// reverse middle dimension: [outer, mid, inner]
__global__ void k_flip_mid(const float* __restrict__ in, float* __restrict__ out,
                           long long outer, int mid, int inner) {
  long long tot = outer * mid * inner;
  long long i = (long long)blockIdx.x * blockDim.x + threadIdx.x;
  if (i >= tot) return;
  int ii = (int)(i % inner);
  long long t = i / inner;
  int m = (int)(t % mid);
  long long o = t / mid;
  out[i] = in[(o * mid + (mid - 1 - m)) * inner + ii];
}

// permute middle dimension: src_m = (m % a)*bd + m/a
__global__ void k_perm_mid(const float* __restrict__ in, float* __restrict__ out,
                           long long outer, int mid, int inner, int a, int bd) {
  long long tot = outer * mid * inner;
  long long i = (long long)blockIdx.x * blockDim.x + threadIdx.x;
  if (i >= tot) return;
  int ii = (int)(i % inner);
  long long t = i / inner;
  int m = (int)(t % mid);
  long long o = t / mid;
  int srcm = (m % a) * bd + (m / a);
  out[i] = in[(o * mid + srcm) * inner + ii];
}

// ---------------------------------------------------------------------------
// InstanceNorm over (H*W) per (b,c): stats + apply (optional relu)
// ---------------------------------------------------------------------------
__global__ void k_inorm_stats(const float* __restrict__ x, float* __restrict__ stats, int L) {
  long long bc = blockIdx.x;
  const float* p = x + bc * L;
  float s = 0.f, s2 = 0.f;
  for (int i = threadIdx.x; i < L; i += 256) { float v = p[i]; s += v; s2 += v * v; }
  __shared__ float rs[256], rq[256];
  rs[threadIdx.x] = s; rq[threadIdx.x] = s2;
  __syncthreads();
  for (int st = 128; st > 0; st >>= 1) {
    if (threadIdx.x < st) { rs[threadIdx.x] += rs[threadIdx.x + st]; rq[threadIdx.x] += rq[threadIdx.x + st]; }
    __syncthreads();
  }
  if (threadIdx.x == 0) {
    float m = rs[0] / L;
    stats[2 * bc] = m;
    stats[2 * bc + 1] = rq[0] / L - m * m;
  }
}
__global__ void k_inorm_apply(const float* __restrict__ x, const float* __restrict__ stats,
                              float* __restrict__ y, int L, int relu, long long total) {
  long long i = (long long)blockIdx.x * blockDim.x + threadIdx.x;
  if (i >= total) return;
  long long bc = i / L;
  float m = stats[2 * bc], v = stats[2 * bc + 1];
  float val = (x[i] - m) * rsqrtf(v + 1e-5f);
  if (relu && val < 0.f) val = 0.f;
  y[i] = val;
}

// ---------------------------------------------------------------------------
// LayerNorm over last dim C of a [rows, C] matrix; one wave per row.
// ---------------------------------------------------------------------------
__global__ void k_layernorm(const float* __restrict__ x, const float* __restrict__ w,
                            const float* __restrict__ b, float* __restrict__ y,
                            int rows, int C) {
  int row = blockIdx.x * (blockDim.x >> 5) + (threadIdx.x >> 5);
  int lane = threadIdx.x & 31;
  if (row >= rows) return;
  const float* p = x + (long long)row * C;
  float s = 0.f, s2 = 0.f;
  for (int c = lane; c < C; c += 32) { float v = p[c]; s += v; s2 += v * v; }
  for (int off = 16; off >= 1; off >>= 1) {
    s  += __shfl_xor(s, off, 32);
    s2 += __shfl_xor(s2, off, 32);
  }
  float m = s / C;
  float r = rsqrtf(s2 / C - m * m + 1e-5f);
  float* q = y + (long long)row * C;
  for (int c = lane; c < C; c += 32) q[c] = (p[c] - m) * r * w[c] + b[c];
}

// ---------------------------------------------------------------------------
// causal depthwise conv1d (k=4, pad-left 3) + bias + SiLU.
// input xz: (B, 2*di, L), channels [0,di) used; output xt: (B, L, di)
// ---------------------------------------------------------------------------
__global__ void k_dwconv_silu(const float* __restrict__ xz, const float* __restrict__ w,
                              const float* __restrict__ bias, float* __restrict__ xt,
                              int Bt, int di, int L) {
  int bd = blockIdx.y;
  int b = bd / di, d = bd % di;
  int l = blockIdx.x * blockDim.x + threadIdx.x;
  if (l >= L) return;
  const float* p = xz + ((long long)b * 2 * di + d) * L;
  float acc = bias[d];
#pragma unroll
  for (int j = 0; j < 4; ++j) {
    int li = l - 3 + j;
    if (li >= 0) acc += w[d * 4 + j] * p[li];
  }
  float sv = acc / (1.f + expf(-acc));
  xt[((long long)b * L + l) * di + d] = sv;
}

// ---------------------------------------------------------------------------
// Selective scan. Wave32: 16 lanes own the 16 SSM states of one channel
// (2 channels per wave). einsum over n via __shfl_xor reduce. Epilogue
// fuses +u*D and *silu(z) (z read transposed from xz channel di+d).
// global_prefetch on the upcoming rows hides the sequential-walk latency.
// ---------------------------------------------------------------------------
__global__ void k_scan(const float* __restrict__ u, const float* __restrict__ delta,
                       const float* __restrict__ dbl, int ldDbl, int dtr,
                       const float* __restrict__ A_log, const float* __restrict__ Dp,
                       const float* __restrict__ xz, float* __restrict__ y,
                       int Bt, int L, int di) {
  int gw = (int)(((long long)blockIdx.x * blockDim.x + threadIdx.x) >> 5);
  int lane = threadIdx.x & 31;
  int ch = gw * 2 + (lane >> 4);
  int n = lane & 15;
  if (ch >= Bt * di) return;              // uniform per wave (Bt*di is even)
  int b = ch / di, d = ch % di;
  float a  = -expf(A_log[d * 16 + n]);
  float Dd = Dp[d];
  const float* zrow = xz + ((long long)b * 2 * di + di + d) * L;
  float h = 0.f;
  for (int l = 0; l < L; ++l) {
    long long r = (long long)b * L + l;
    if ((l & 31) == 0 && l + 32 < L) {
      long long rp = r + 32;
      __builtin_prefetch(&u[rp * di + d], 0, 1);
      __builtin_prefetch(&delta[rp * di + d], 0, 1);
      __builtin_prefetch(&dbl[rp * ldDbl + dtr + n], 0, 1);
    }
    float ul = u[r * di + d];
    float dt = delta[r * di + d];
    float Bn = dbl[r * ldDbl + dtr + n];
    float Cn = dbl[r * ldDbl + dtr + 16 + n];
    h = expf(dt * a) * h + dt * ul * Bn;
    float contrib = h * Cn;
    contrib += __shfl_xor(contrib, 1, 32);
    contrib += __shfl_xor(contrib, 2, 32);
    contrib += __shfl_xor(contrib, 4, 32);
    contrib += __shfl_xor(contrib, 8, 32);
    if (n == 0) {
      float z = zrow[l];
      float yv = contrib + ul * Dd;
      y[r * di + d] = yv * (z / (1.f + expf(-z)));
    }
  }
}

// ===========================================================================
// Host-side orchestration
// ===========================================================================
namespace {

struct Arena {
  char* base; size_t off; size_t cap;
  float* alloc(long long nfloats) {
    size_t bytes = ((size_t)nfloats * 4 + 255) & ~(size_t)255;
    float* p = (float*)(base + off);
    off += bytes;
    return p;
  }
};

struct BranchP { const float *A_log, *D, *conv_b, *conv_w, *dt_b, *dt_w, *xproj_w; };
struct LayerP  { BranchP bwd, fwd; const float *in_w, *ln_b, *ln_w, *out_w; BranchP slc; };
struct Params {
  const float* down_b[3]; const float* down_w[3];
  const float* gsc[4][8];   // p1_b,p1_w,p2_b,p2_w,p3_b,p3_w,p4_b,p4_w
  const float* mlp[4][4];   // fc1_b,fc1_w,fc2_b,fc2_w
  LayerP stages[4][2];
  const float *stem_b, *stem_w;
};

// Assumed flattening: d_in[0]=x, then params leaves with dicts in sorted-key
// order (jax tree order): down, gsc, mlp, stages, stem_b, stem_w.
void load_params(void* const* d_in, Params& P) {
  int i = 1;
  auto F = [&]() -> const float* { return (const float*)d_in[i++]; };
  for (int k = 0; k < 3; ++k) { P.down_b[k] = F(); P.down_w[k] = F(); }
  for (int k = 0; k < 4; ++k) for (int j = 0; j < 8; ++j) P.gsc[k][j] = F();
  for (int k = 0; k < 4; ++k) for (int j = 0; j < 4; ++j) P.mlp[k][j] = F();
  auto BR = [&](BranchP& br) {
    br.A_log = F(); br.D = F(); br.conv_b = F(); br.conv_w = F();
    br.dt_b = F(); br.dt_w = F(); br.xproj_w = F();
  };
  for (int st = 0; st < 4; ++st)
    for (int ly = 0; ly < 2; ++ly) {
      LayerP& Lp = P.stages[st][ly];
      BR(Lp.bwd); BR(Lp.fwd);
      Lp.in_w = F(); Lp.ln_b = F(); Lp.ln_w = F(); Lp.out_w = F();
      BR(Lp.slc);
    }
  P.stem_b = F(); P.stem_w = F();
}

inline int blks(long long n) { return (int)((n + 255) / 256); }

void launch_gemm(hipStream_t s, const float* A, int lda, const float* W, int ldw,
                 const float* bias, const float* resid, float* C,
                 int M, int N, int K, int act) {
  dim3 g((N + 63) / 64, (M + 63) / 64);
  k_gemm<<<g, 256, 0, s>>>(A, lda, W, ldw, bias, resid, C, M, N, K, act);
}

void inorm(hipStream_t s, Arena& ar, const float* x, float* out,
           int Bt, int C, int L, int relu) {
  size_t mark = ar.off;
  float* stats = ar.alloc((long long)2 * Bt * C);
  k_inorm_stats<<<Bt * C, 256, 0, s>>>(x, stats, L);
  long long tot = (long long)Bt * C * L;
  k_inorm_apply<<<blks(tot), 256, 0, s>>>(x, stats, out, L, relu, tot);
  ar.off = mark;
}

void conv2d_scratch(hipStream_t s, Arena& ar, const float* xin,
                    int Bt, int Cin, int H, int W,
                    const float* w, const float* bias,
                    int KH, int KW, int stride, int pad, int Cout,
                    float* outNCHW) {
  size_t mark = ar.off;
  int OH = (H + 2 * pad - KH) / stride + 1;
  int OW = (W + 2 * pad - KW) / stride + 1;
  long long M = (long long)Bt * OH * OW;
  int K = Cin * KH * KW;
  float* col = ar.alloc(M * K);
  k_im2col<<<blks(M * K), 256, 0, s>>>(xin, col, Bt, Cin, H, W, KH, KW, stride, pad, OH, OW);
  float* gout = ar.alloc(M * Cout);
  launch_gemm(s, col, K, w, K, bias, nullptr, gout, (int)M, Cout, K, ACT_NONE);
  k_blc2bcl<<<blks(M * Cout), 256, 0, s>>>(gout, nullptr, outNCHW, Bt, OH * OW, Cout);
  ar.off = mark;
}

void gsc_block(hipStream_t s, Arena& ar, float* x, const float* const* pp,
               int Bt, int C, int Hh, int Ww) {
  size_t mark = ar.off;
  int L = Hh * Ww;
  long long sz = (long long)Bt * C * L;
  float* t1 = ar.alloc(sz);
  float* t2 = ar.alloc(sz);
  conv2d_scratch(s, ar, x, Bt, C, Hh, Ww, pp[1], pp[0], 3, 3, 1, 1, C, t1);
  inorm(s, ar, t1, t1, Bt, C, L, 1);
  conv2d_scratch(s, ar, t1, Bt, C, Hh, Ww, pp[3], pp[2], 3, 3, 1, 1, C, t1);
  inorm(s, ar, t1, t1, Bt, C, L, 1);
  conv2d_scratch(s, ar, x, Bt, C, Hh, Ww, pp[5], pp[4], 1, 1, 1, 0, C, t2);
  inorm(s, ar, t2, t2, Bt, C, L, 1);
  k_add2<<<blks(sz), 256, 0, s>>>(t1, t2, t1, sz);
  conv2d_scratch(s, ar, t1, Bt, C, Hh, Ww, pp[7], pp[6], 1, 1, 1, 0, C, t2);
  inorm(s, ar, t2, t2, Bt, C, L, 1);
  k_add2<<<blks(sz), 256, 0, s>>>(t2, x, x, sz);
  ar.off = mark;
}

void mamba_branch(hipStream_t s, Arena& ar, const float* xz, const BranchP& p,
                  int Bt, int L, int di, int dtr, float* y) {
  size_t mark = ar.off;
  float* xt = ar.alloc((long long)Bt * L * di);
  dim3 gb((L + 255) / 256, Bt * di);
  k_dwconv_silu<<<gb, 256, 0, s>>>(xz, p.conv_w, p.conv_b, xt, Bt, di, L);
  int Ndbl = dtr + 32;
  float* dbl = ar.alloc((long long)Bt * L * Ndbl);
  launch_gemm(s, xt, di, p.xproj_w, di, nullptr, nullptr, dbl, Bt * L, Ndbl, di, ACT_NONE);
  float* delta = ar.alloc((long long)Bt * L * di);
  launch_gemm(s, dbl, Ndbl, p.dt_w, dtr, p.dt_b, nullptr, delta, Bt * L, di, dtr, ACT_SOFTPLUS);
  int waves = (Bt * di + 1) / 2;
  k_scan<<<blks((long long)waves * 32), 256, 0, s>>>(
      xt, delta, dbl, Ndbl, dtr, p.A_log, p.D, xz, y, Bt, L, di);
  ar.off = mark;
}

void mamba_layer(hipStream_t s, Arena& ar, float* x, const LayerP& p,
                 int Bt, int C, int Hh, int Ww, int nsl) {
  int L = Hh * Ww;
  int di = 2 * C;
  int dtr = (C + 15) / 16;
  size_t mark = ar.off;

  float* xf = ar.alloc((long long)Bt * L * C);
  k_bcl2blc<<<blks((long long)Bt * C * L), 256, 0, s>>>(x, xf, Bt, L, C);
  float* xn = ar.alloc((long long)Bt * L * C);
  int rowsLN = Bt * L;
  k_layernorm<<<(rowsLN + 7) / 8, 256, 0, s>>>(xf, p.ln_w, p.ln_b, xn, rowsLN, C);

  float* xzb = ar.alloc((long long)Bt * L * 2 * di);
  launch_gemm(s, xn, C, p.in_w, C, nullptr, nullptr, xzb, Bt * L, 2 * di, C, ACT_NONE);
  long long totxz = (long long)Bt * 2 * di * L;
  float* xz = ar.alloc(totxz);
  k_blc2bcl<<<blks(totxz), 256, 0, s>>>(xzb, nullptr, xz, Bt, L, 2 * di);

  long long toty = (long long)Bt * L * di;
  float* yf = ar.alloc(toty);
  float* yb = ar.alloc(toty);
  float* ys = ar.alloc(toty);

  mamba_branch(s, ar, xz, p.fwd, Bt, L, di, dtr, yf);

  { // backward: flip along L, branch, flip result back
    size_t m2 = ar.off;
    float* xzf = ar.alloc(totxz);
    k_flip_mid<<<blks(totxz), 256, 0, s>>>(xz, xzf, (long long)Bt * 2 * di, L, 1);
    float* ybr = ar.alloc(toty);
    mamba_branch(s, ar, xzf, p.bwd, Bt, L, di, dtr, ybr);
    k_flip_mid<<<blks(toty), 256, 0, s>>>(ybr, yb, Bt, L, di);
    ar.off = m2;
  }
  { // slice branch: interleave permutation of L, branch, inverse permutation
    size_t m2 = ar.off;
    int Ln = L / nsl;
    float* xzp = ar.alloc(totxz);
    k_perm_mid<<<blks(totxz), 256, 0, s>>>(xz, xzp, (long long)Bt * 2 * di, L, 1, nsl, Ln);
    float* ysr = ar.alloc(toty);
    mamba_branch(s, ar, xzp, p.slc, Bt, L, di, dtr, ysr);
    k_perm_mid<<<blks(toty), 256, 0, s>>>(ysr, ys, Bt, L, di, Ln, nsl);
    ar.off = m2;
  }

  k_add3<<<blks(toty), 256, 0, s>>>(yf, yb, ys, yf, toty);
  float* outb = ar.alloc((long long)Bt * L * C);
  launch_gemm(s, yf, di, p.out_w, di, nullptr, nullptr, outb, Bt * L, C, di, ACT_NONE);
  // transpose to NCHW + residual (in-place safe: elementwise same index)
  k_blc2bcl<<<blks((long long)Bt * C * L), 256, 0, s>>>(outb, x, x, Bt, L, C);
  ar.off = mark;
}

} // namespace

extern "C" void kernel_launch(void* const* d_in, const int* in_sizes, int n_in,
                              void* d_out, int out_size, void* d_ws, size_t ws_size,
                              hipStream_t stream) {
  (void)n_in; (void)out_size;
  Params P;
  load_params(d_in, P);
  const float* xin = (const float*)d_in[0];
  const int Bt = in_sizes[0] / (192 * 192);   // IN_CHANS=1, H=W=192

  Arena ar{(char*)d_ws, 0, ws_size};
  float* out = (float*)d_out;

  static const int DIMSa[4] = {48, 96, 192, 384};
  static const int NSL[4]   = {32, 16, 8, 4};

  float* x = ar.alloc((long long)Bt * 48 * 96 * 96);   // persistent activation
  long long outoff = 0;

  for (int st = 0; st < 4; ++st) {
    int C = DIMSa[st];
    int Hc = 96 >> st, Wc = 96 >> st;
    int L = Hc * Wc;

    if (st == 0) {
      size_t mark = ar.off;
      conv2d_scratch(stream, ar, xin, Bt, 1, 192, 192, P.stem_w, P.stem_b,
                     7, 7, 2, 3, 48, x);
      ar.off = mark;
    } else {
      int Cp = DIMSa[st - 1], Hp = 96 >> (st - 1);
      size_t mark = ar.off;
      float* tmp = ar.alloc((long long)Bt * Cp * Hp * Hp);
      inorm(stream, ar, x, tmp, Bt, Cp, Hp * Hp, 0);
      conv2d_scratch(stream, ar, tmp, Bt, Cp, Hp, Hp,
                     P.down_w[st - 1], P.down_b[st - 1], 2, 2, 2, 0, C, x);
      ar.off = mark;
    }

    gsc_block(stream, ar, x, P.gsc[st], Bt, C, Hc, Wc);

    for (int ly = 0; ly < 2; ++ly)
      mamba_layer(stream, ar, x, P.stages[st][ly], Bt, C, Hc, Wc, NSL[st]);

    { // MLP head -> d_out slice
      size_t mark = ar.off;
      long long sz = (long long)Bt * C * L;
      float* xo = ar.alloc(sz);
      inorm(stream, ar, x, xo, Bt, C, L, 0);
      float* xb = ar.alloc((long long)Bt * L * C);
      k_bcl2blc<<<blks(sz), 256, 0, stream>>>(xo, xb, Bt, L, C);
      float* h = ar.alloc((long long)Bt * L * 2 * C);
      launch_gemm(stream, xb, C, P.mlp[st][1], C, P.mlp[st][0], nullptr,
                  h, Bt * L, 2 * C, C, ACT_GELU);
      float* o = ar.alloc((long long)Bt * L * C);
      launch_gemm(stream, h, 2 * C, P.mlp[st][3], 2 * C, P.mlp[st][2], nullptr,
                  o, Bt * L, C, 2 * C, ACT_NONE);
      k_blc2bcl<<<blks(sz), 256, 0, stream>>>(o, nullptr, out + outoff, Bt, L, C);
      outoff += sz;
      ar.off = mark;
    }
  }
}
